// SpatialAPPNP_89653147337199
// MI455X (gfx1250) — compile-verified, matrix operand order checked by reading, not run on hardware
//
#include <hip/hip_runtime.h>
#include <hip/hip_bf16.h>
#include <math.h>

typedef __attribute__((ext_vector_type(16))) _Float16 v16h;
typedef __attribute__((ext_vector_type(8)))  float    v8f;

#define J_   17
#define C_   128
#define NEGV -1e9f

// adjacency bitmasks (17 joints, symmetric + self)
__device__ __constant__ unsigned int ADJM[17] = {
    0x00093u, 0x00007u, 0x0000Eu, 0x0000Cu, 0x00031u, 0x00070u, 0x00060u,
    0x00181u, 0x04B80u, 0x00700u, 0x00600u, 0x01900u, 0x03800u, 0x03000u,
    0x0C100u, 0x1C000u, 0x18000u};

__device__ __forceinline__ float wave_red(float v) {
#pragma unroll
    for (int m = 16; m > 0; m >>= 1) v += __shfl_xor(v, m, 32);
    return v;
}

// ---- weight pre-pass: f32 [K][N] -> f16 [N][K] (K contiguous) into d_ws ----
// layout in ws (halves): w1t [256][128] @0, w2t [256][128] @32768, vwt [128][128] @65536
__global__ __launch_bounds__(256) void convert_weights_k(
    const float* __restrict__ w1, const float* __restrict__ w2,
    const float* __restrict__ vw, _Float16* __restrict__ ws) {
    int idx = blockIdx.x * 256 + threadIdx.x;
    if (idx < 32768) {
        int n = idx >> 7, k = idx & 127;
        ws[idx] = (_Float16)w1[k * 256 + n];
    } else if (idx < 65536) {
        int t = idx - 32768; int n = t >> 7, k = t & 127;
        ws[idx] = (_Float16)w2[k * 256 + n];
    } else if (idx < 81920) {
        int t = idx - 65536; int n = t >> 7, k = t & 127;
        ws[idx] = (_Float16)vw[k * 128 + n];
    }
}

// ---- LDS partition (floats / halves), total 59,856 bytes ----
// consts: og 0, ob 128, lg 256, lb 384, aw 512, b1 640(256), b2 896(256),
//         vb 1152(128), psw 1280(128), scalars 1408: [0]=alpha_b [1]=ps_b
#define CONST_F 1424

__global__ __launch_bounds__(256) void spatial_gat_k(
    const float* __restrict__ x,
    const float* __restrict__ og, const float* __restrict__ ob,
    const float* __restrict__ lg, const float* __restrict__ lb,
    const float* __restrict__ aw, const float* __restrict__ ab,
    const float* __restrict__ b1, const float* __restrict__ b2,
    const float* __restrict__ psw, const float* __restrict__ psb,
    const float* __restrict__ vb,
    const _Float16* __restrict__ wt,
    float* __restrict__ out) {
    extern __shared__ __align__(16) char smem[];
    float*     cst  = (float*)smem;               // 1424 f
    float*     xf   = cst + CONST_F;              // 17*128 f  (current state)
    float*     x0   = xf + 2176;                  // 17*128 f
    _Float16*  xnh  = (_Float16*)(x0 + 2176);     // 32*128 h  (padded A tile)
    _Float16*  lbuf = xnh + 32 * 128;             // 17*256 h
    _Float16*  rbuf = lbuf + 17 * 256;            // 17*256 h
    float*     vbuf = (float*)(rbuf + 17 * 256);  // 17*128 f
    float*     sbuf = vbuf + 2176;                // 2*17*17 f (pad 580)
    float*     skv  = sbuf + 580;                 // 17 f (pad 32)

    const int tid  = threadIdx.x;
    const int wave = tid >> 5;
    const int lane = tid & 31;

    // ---- load constants into LDS ----
    if (tid < 128) {
        int i = tid;
        cst[i]        = og[i];  cst[128 + i]  = ob[i];
        cst[256 + i]  = lg[i];  cst[384 + i]  = lb[i];
        cst[512 + i]  = aw[i];  cst[1152 + i] = vb[i];
        cst[1280 + i] = psw[i];
    }
    for (int i = tid; i < 256; i += 256) { cst[640 + i] = b1[i]; cst[896 + i] = b2[i]; }
    if (tid == 0) { cst[1408] = ab[0]; cst[1409] = psb[0]; }
    // zero padded A rows 17..31 (stay zero for the whole kernel)
    for (int i = tid; i < (32 - J_) * C_; i += 256) xnh[J_ * C_ + i] = (_Float16)0.f;
    // warm L2 with the f16 weight buffers (global_prefetch_b8)
    for (int off = tid * 64; off < 81920; off += 256 * 64)
        __builtin_prefetch(wt + off, 0, 1);

    __syncthreads();

    // ---- outer LN: x = LN(x_in)*og+ob ; keep as xf and x0 ----
    const float* xin = x + (size_t)blockIdx.x * (J_ * C_);
    for (int j = wave; j < J_; j += 8) {
        float4 v = *(const float4*)(xin + j * C_ + lane * 4);
        float s  = v.x + v.y + v.z + v.w;
        float s2 = v.x * v.x + v.y * v.y + v.z * v.z + v.w * v.w;
        s = wave_red(s); s2 = wave_red(s2);
        float m   = s * (1.f / 128.f);
        float inv = rsqrtf(s2 * (1.f / 128.f) - m * m + 1e-5f);
        float4 g  = *(const float4*)(cst + 0 + lane * 4);
        float4 bb = *(const float4*)(cst + 128 + lane * 4);
        float4 r;
        r.x = (v.x - m) * inv * g.x + bb.x;
        r.y = (v.y - m) * inv * g.y + bb.y;
        r.z = (v.z - m) * inv * g.z + bb.z;
        r.w = (v.w - m) * inv * g.w + bb.w;
        *(float4*)(xf + j * C_ + lane * 4) = r;
        *(float4*)(x0 + j * C_ + lane * 4) = r;
    }
    __syncthreads();

    for (int it = 0; it < 4; ++it) {
        // ---- inner LN -> xnh (f16 A tile) ; skip gate via alpha dot ----
        for (int j = wave; j < J_; j += 8) {
            float4 v = *(const float4*)(xf + j * C_ + lane * 4);
            float s  = v.x + v.y + v.z + v.w;
            float s2 = v.x * v.x + v.y * v.y + v.z * v.z + v.w * v.w;
            s = wave_red(s); s2 = wave_red(s2);
            float m   = s * (1.f / 128.f);
            float inv = rsqrtf(s2 * (1.f / 128.f) - m * m + 1e-5f);
            float4 g  = *(const float4*)(cst + 256 + lane * 4);
            float4 bb = *(const float4*)(cst + 384 + lane * 4);
            float n0 = (v.x - m) * inv * g.x + bb.x;
            float n1 = (v.y - m) * inv * g.y + bb.y;
            float n2 = (v.z - m) * inv * g.z + bb.z;
            float n3 = (v.w - m) * inv * g.w + bb.w;
            float4 a4 = *(const float4*)(cst + 512 + lane * 4);
            float ad  = wave_red(n0 * a4.x + n1 * a4.y + n2 * a4.z + n3 * a4.w);
            if (lane == 0) skv[j] = 1.f / (1.f + expf(-(ad + cst[1408])));
            _Float16* p = xnh + j * C_ + lane * 4;
            p[0] = (_Float16)n0; p[1] = (_Float16)n1;
            p[2] = (_Float16)n2; p[3] = (_Float16)n3;
        }
        __syncthreads();

        // ---- WMMA GEMMs: l = xn@w1+b1, r = xn@w2+b2, v = xn@v_w+v_b ----
        union AF { v16h v; uint4 q[2]; };
        const int mrow = lane & 15;
        const int hi   = lane >> 4;     // lanes 16-31 hold the +8 K groups
        AF af[2][4];
#pragma unroll
        for (int mt = 0; mt < 2; ++mt)
#pragma unroll
            for (int kt = 0; kt < 4; ++kt) {
                const _Float16* pa = xnh + (mt * 16 + mrow) * C_ + kt * 32 + hi * 8;
                af[mt][kt].q[0] = *(const uint4*)pa;        // K: base+0..7
                af[mt][kt].q[1] = *(const uint4*)(pa + 16); // K: base+16..23
            }
#pragma unroll
        for (int t = 0; t < 5; ++t) {
            int nt = wave + t * 8;  // 0..39 over [w1(16) | w2(16) | vw(8)]
            const _Float16* wb; const float* bias; int nloc;
            if (nt < 16)      { wb = wt;         bias = cst + 640;  nloc = nt; }
            else if (nt < 32) { wb = wt + 32768; bias = cst + 896;  nloc = nt - 16; }
            else              { wb = wt + 65536; bias = cst + 1152; nloc = nt - 32; }
            const int ncol = nloc * 16 + mrow;
            v8f a0 = {}, a1 = {};
#pragma unroll
            for (int kt = 0; kt < 4; ++kt) {
                AF bf;
                const _Float16* pb = wb + ncol * 128 + kt * 32 + hi * 8;
                bf.q[0] = *(const uint4*)pb;
                bf.q[1] = *(const uint4*)(pb + 16);
                a0 = __builtin_amdgcn_wmma_f32_16x16x32_f16(
                         false, af[0][kt].v, false, bf.v, (short)0, a0, false, false);
                a1 = __builtin_amdgcn_wmma_f32_16x16x32_f16(
                         false, af[1][kt].v, false, bf.v, (short)0, a1, false, false);
            }
            float bv = bias[ncol];
            if (nt < 32) {  // l / r, stored f16 [17][256]
                _Float16* dsth = (nt < 16) ? lbuf : rbuf;
#pragma unroll
                for (int r = 0; r < 8; ++r) {
                    int m = hi ? 8 + r : r;                 // rows 0..15
                    dsth[m * 256 + ncol] = (_Float16)(a0[r] + bv);
                }
                if (!hi) dsth[16 * 256 + ncol] = (_Float16)(a1[0] + bv); // row 16
            } else {        // v, stored f32 [17][128]
#pragma unroll
                for (int r = 0; r < 8; ++r) {
                    int m = hi ? 8 + r : r;
                    vbuf[m * 128 + ncol] = a0[r] + bv;
                }
                if (!hi) vbuf[16 * 128 + ncol] = a1[0] + bv;
            }
        }
        __syncthreads();

        // ---- attention scores: s[h,i,j] = sum_d psw[d]*lrelu(l[i,h,d]+r[j,h,d]) ----
        const float psb_ = cst[1409];
        for (int idx = tid; idx < 2 * J_ * J_; idx += 256) {
            int h = idx / (J_ * J_);
            int rem = idx - h * (J_ * J_);
            int i = rem / J_, jj = rem - i * J_;
            float sv = NEGV;
            if (ADJM[i] & (1u << jj)) {
                const _Float16* lr = lbuf + i * 256 + h * 128;
                const _Float16* rr = rbuf + jj * 256 + h * 128;
                const float*    pw = cst + 1280;
                const int start = (jj * 8) & 127;  // de-conflict LDS banks
                float acc = 0.f;
                for (int dd = 0; dd < 128; ++dd) {
                    int d = (dd + start) & 127;
                    float z = (float)lr[d] + (float)rr[d];
                    z = z > 0.f ? z : 0.2f * z;
                    acc += pw[d] * z;
                }
                sv = acc + psb_;
            }
            sbuf[(h * J_ + i) * J_ + jj] = sv;
        }
        __syncthreads();

        // ---- softmax over j for each (h,i) row ----
        if (tid < 2 * J_) {
            float* row = sbuf + tid * J_;
            float mx = NEGV;
            for (int j = 0; j < J_; ++j) mx = fmaxf(mx, row[j]);
            float sum = 0.f;
            for (int j = 0; j < J_; ++j) { float e = expf(row[j] - mx); row[j] = e; sum += e; }
            float inv = 1.f / sum;
            for (int j = 0; j < J_; ++j) row[j] *= inv;
        }
        __syncthreads();

        // ---- o = gelu(S @ V) ; x = (1-skip)*o + skip*x0 ----
        for (int idx = tid; idx < J_ * C_; idx += 256) {
            int i = idx >> 7, c = idx & 127, h = c >> 6;
            const float* srow = sbuf + (h * J_ + i) * J_;
            float acc = 0.f;
#pragma unroll
            for (int jj = 0; jj < J_; ++jj) acc += srow[jj] * vbuf[jj * C_ + c];
            float g  = 0.5f * acc * (1.f + erff(acc * 0.70710678f));
            float sk = skv[i];
            xf[idx] = (1.f - sk) * g + sk * x0[idx];
        }
        __syncthreads();
    }

    // ---- write result ----
    float* orow = out + (size_t)blockIdx.x * (J_ * C_);
    for (int q = tid; q < (J_ * C_) / 4; q += 256)
        *(float4*)(orow + q * 4) = *(const float4*)(xf + q * 4);
}

extern "C" void kernel_launch(void* const* d_in, const int* in_sizes, int n_in,
                              void* d_out, int out_size, void* d_ws, size_t ws_size,
                              hipStream_t stream) {
    (void)n_in; (void)out_size; (void)ws_size;
    const float* x   = (const float*)d_in[0];
    const float* og  = (const float*)d_in[1];
    const float* ob  = (const float*)d_in[2];
    const float* lg  = (const float*)d_in[3];
    const float* lb  = (const float*)d_in[4];
    const float* aw  = (const float*)d_in[5];
    const float* ab  = (const float*)d_in[6];
    const float* w1  = (const float*)d_in[7];
    const float* b1  = (const float*)d_in[8];
    const float* w2  = (const float*)d_in[9];
    const float* b2  = (const float*)d_in[10];
    const float* psw = (const float*)d_in[11];
    const float* psb = (const float*)d_in[12];
    const float* vw  = (const float*)d_in[13];
    const float* vb  = (const float*)d_in[14];
    _Float16* wt = (_Float16*)d_ws;  // 81920 halves = 160 KB

    convert_weights_k<<<320, 256, 0, stream>>>(w1, w2, vw, wt);

    const int nbt = in_sizes[0] / (J_ * C_);  // 16*243 = 3888 groups
    const size_t lds_bytes =
        (CONST_F + 2176 + 2176) * sizeof(float) +      // consts, xf, x0
        (32 * 128 + 17 * 256 + 17 * 256) * sizeof(_Float16) +  // xnh, l, r
        (2176 + 580 + 32) * sizeof(float);             // v, scores, skip
    spatial_gat_k<<<nbt, 256, lds_bytes, stream>>>(
        x, og, ob, lg, lb, aw, ab, b1, b2, psw, psb, vb, wt, (float*)d_out);
}
